// cond_RNN_LSTM_embed_58248346468905
// MI455X (gfx1250) — matrix operand order repaired and checked by reading, not compiled, not run on hardware
//
#include <hip/hip_runtime.h>
#include <hip/hip_bf16.h>

// ---------------------------------------------------------------------------
// cond RNN/LSTM + embed + big FC for MI455X (gfx1250), wave32 WMMA f16->f32.
//
// Shapes: B=16, T=1024, E=128 (2E=256), H=256 (4H=1024), C=8000.
//   inp   = concat(embed_x[x0], embed_l[x1])           (B*T, 256)  f16
//   xproj = inp @ W_ih^T + b_ih + b_hh  -> (t,b,4H)    f32
//   scan  : gates = xproj[t] + h @ W_hh^T ; LSTM cell  (persistent 1 block)
//   out   = hs @ fc_w^T + fc_b          -> (b,t,C)     f32 (512 MB)
//
// All GEMMs are A(M,K) * W(N,K)^T with K-contiguous rows, mapping directly
// onto the gfx1250 16x16x32 f16 WMMA per-lane fragment layouts (contiguous
// 16B/32B chunks). Fragment loads are software-pipelined one k-step ahead so
// v_wmma executes with the next loads already in flight.
// ---------------------------------------------------------------------------

typedef __attribute__((ext_vector_type(16))) _Float16 v16h;
typedef __attribute__((ext_vector_type(8)))  _Float16 v8h;
typedef __attribute__((ext_vector_type(8)))  float    v8f;

union AFragU { v16h v; v8h h[2]; };

// A fragment: 16x32 f16, rows mB..mB+15, K = kB..kB+31, row-major src (ld halves)
static __device__ __forceinline__ v16h load_a_frag(const _Float16* A, int ld,
                                                   int mB, int kB, int lane) {
  int row = mB + (lane & 15);
  int kk  = kB + (lane >> 4) * 8;
  const _Float16* p = A + row * ld + kk;
  AFragU u;
  u.h[0] = *reinterpret_cast<const v8h*>(p);        // K = kk .. kk+7
  u.h[1] = *reinterpret_cast<const v8h*>(p + 16);   // K = kk+16 .. kk+23
  return u.v;
}

// B fragment: 32x16 f16 from N-major weight (N x K row-major, computing A*W^T)
static __device__ __forceinline__ v16h load_b_frag(const _Float16* W, int ld,
                                                   int nB, int kB, int lane) {
  int col = nB + (lane & 15);
  int kk  = kB + (lane >> 4) * 8;
  const _Float16* p = W + col * ld + kk;
  AFragU u;
  u.h[0] = *reinterpret_cast<const v8h*>(p);        // K = kk .. kk+7
  u.h[1] = *reinterpret_cast<const v8h*>(p + 16);   // K = kk+16 .. kk+23
  return u.v;
}

static __device__ __forceinline__ v8f wmma_f16(v16h a, v16h b, v8f c) {
  return __builtin_amdgcn_wmma_f32_16x16x32_f16(false, a, false, b,
                                                (short)0, c, false, false);
}

static __device__ __forceinline__ float sigmoidf_(float x) {
  return 1.0f / (1.0f + __expf(-x));
}

// ---------------------------------------------------------------------------
// K1: f32 -> f16 convert (grid-stride)
// ---------------------------------------------------------------------------
__global__ __launch_bounds__(256) void cvt_f32_f16(const float* __restrict__ src,
                                                   _Float16* __restrict__ dst, int n) {
  for (int i = blockIdx.x * blockDim.x + threadIdx.x; i < n;
       i += gridDim.x * blockDim.x)
    dst[i] = (_Float16)src[i];
}

// ---------------------------------------------------------------------------
// K2: embedding gather + concat -> inp_f16[(b*T+t), 256]
// ---------------------------------------------------------------------------
__global__ __launch_bounds__(256) void embed_gather(const long long* __restrict__ x,
                                                    const float* __restrict__ ew,
                                                    const float* __restrict__ lw,
                                                    _Float16* __restrict__ inp) {
  int m = blockIdx.x;            // m = b*T + t  (16384 blocks)
  int e = threadIdx.x;           // 0..255
  long long x0 = x[(size_t)m * 2 + 0];
  long long x1 = x[(size_t)m * 2 + 1];
  float v = (e < 128) ? ew[(size_t)x0 * 128 + e]
                      : lw[(size_t)x1 * 128 + (e - 128)];
  inp[(size_t)m * 256 + e] = (_Float16)v;
}

// ---------------------------------------------------------------------------
// K3: input projection GEMM: xproj[t,b,g] = inp @ W_ih^T + b_ih + b_hh
//     M=16384, N=1024, K=256.  grid (1024, 2), 256 thr = 8 waves x 4 n-tiles.
//     Fragments double-buffered one k-step ahead.
// ---------------------------------------------------------------------------
__global__ __launch_bounds__(256) void in_proj(const _Float16* __restrict__ inp,
                                               const _Float16* __restrict__ wih,
                                               const float* __restrict__ bih,
                                               const float* __restrict__ bhh,
                                               float* __restrict__ xproj) {
  int wave = threadIdx.x >> 5, lane = threadIdx.x & 31;
  int hf = lane >> 4, l15 = lane & 15;
  int mB = blockIdx.x * 16;
  int nBase = blockIdx.y * 512 + wave * 64;

  v8f acc[4] = {};
  v16h aCur = load_a_frag(inp, 256, mB, 0, lane);
  v16h bCur[4];
#pragma unroll
  for (int t = 0; t < 4; ++t)
    bCur[t] = load_b_frag(wih, 256, nBase + t * 16, 0, lane);

#pragma unroll
  for (int k = 0; k < 8; ++k) {
    v16h aNxt;
    v16h bNxt[4];
    if (k < 7) {                                  // prefetch k+1 (compile-time)
      aNxt = load_a_frag(inp, 256, mB, (k + 1) * 32, lane);
#pragma unroll
      for (int t = 0; t < 4; ++t)
        bNxt[t] = load_b_frag(wih, 256, nBase + t * 16, (k + 1) * 32, lane);
    }
#pragma unroll
    for (int t = 0; t < 4; ++t) acc[t] = wmma_f16(aCur, bCur[t], acc[t]);
    if (k < 7) {
      aCur = aNxt;
#pragma unroll
      for (int t = 0; t < 4; ++t) bCur[t] = bNxt[t];
    }
  }

#pragma unroll
  for (int t = 0; t < 4; ++t) {
    int g = nBase + t * 16 + l15;
    float bias = bih[g] + bhh[g];
#pragma unroll
    for (int v = 0; v < 8; ++v) {
      int m = mB + v + hf * 8;
      int b = m >> 10, tt = m & 1023;             // m = b*T + t
      xproj[((size_t)tt * 16 + b) * 1024 + g] = acc[t][v] + bias;
    }
  }
}

// ---------------------------------------------------------------------------
// K4: persistent LSTM scan. 1 block, 256 thr = 8 waves. h,c,gates in LDS.
//     Per step: gates(16x1024) = xproj[t] + h(16x256) @ W_hh^T, 8 waves x
//     8 n-tiles x 8 k-steps of v_wmma_f32_16x16x32_f16, then LSTM cell.
//     A fragments (h) hoisted: 8 ds-fragment loads per step, reused x8 tiles.
// ---------------------------------------------------------------------------
__global__ __launch_bounds__(256) void lstm_scan(const float* __restrict__ xproj,
                                                 const _Float16* __restrict__ whh,
                                                 _Float16* __restrict__ hs) {
  __shared__ __align__(32) _Float16 hLds[16][256];   //  8 KB
  __shared__ __align__(32) float    cLds[16][256];   // 16 KB
  __shared__ __align__(32) _Float16 gLds[16][1024];  // 32 KB

  int tid = threadIdx.x;
  int wave = tid >> 5, lane = tid & 31;
  int hf = lane >> 4, l15 = lane & 15;

  for (int i = tid; i < 16 * 256; i += 256) {
    (&hLds[0][0])[i] = (_Float16)0.0f;
    (&cLds[0][0])[i] = 0.0f;
  }
  __syncthreads();

  for (int t = 0; t < 1024; ++t) {
    // ---- A fragments of h, loaded once per step from LDS ----
    v16h a[8];
#pragma unroll
    for (int k = 0; k < 8; ++k)
      a[k] = load_a_frag(&hLds[0][0], 256, 0, k * 32, lane);

    // ---- gates = h @ W_hh^T + xproj[t] ----
    for (int nt = 0; nt < 8; ++nt) {
      int nB = (wave * 8 + nt) * 16;
      v8f acc = {};
      v16h bCur = load_b_frag(whh, 256, nB, 0, lane);
#pragma unroll
      for (int k = 0; k < 8; ++k) {
        v16h bNxt;
        if (k < 7) bNxt = load_b_frag(whh, 256, nB, (k + 1) * 32, lane);
        acc = wmma_f16(a[k], bCur, acc);
        if (k < 7) bCur = bNxt;
      }
      int g = nB + l15;
      const float* xp = xproj + (size_t)t * 16384 + g;
#pragma unroll
      for (int v = 0; v < 8; ++v) {
        int brow = v + hf * 8;
        gLds[brow][g] = (_Float16)(acc[v] + xp[(size_t)brow * 1024]);
      }
    }
    __syncthreads();

    // ---- elementwise LSTM cell: 16x256 items over 256 threads ----
    for (int it = tid; it < 4096; it += 256) {
      int b = it >> 8, j = it & 255;
      float gi = sigmoidf_((float)gLds[b][j]);
      float gf = sigmoidf_((float)gLds[b][256 + j]);
      float gg = tanhf((float)gLds[b][512 + j]);
      float go = sigmoidf_((float)gLds[b][768 + j]);
      float cn = gf * cLds[b][j] + gi * gg;
      float hn = go * tanhf(cn);
      cLds[b][j] = cn;
      _Float16 h16 = (_Float16)hn;
      hLds[b][j] = h16;
      hs[((size_t)b * 1024 + t) * 256 + j] = h16;   // hs[(b*T+t), H]
    }
    __syncthreads();
  }
}

// ---------------------------------------------------------------------------
// K5: FC GEMM: out[m, c] = hs @ fc_w^T + fc_b.  M=16384, N=8000, K=256.
//     grid (1024, 16), 256 thr = 8 waves x 4 n-tiles, double-buffered frags.
//     Tail tiles (>=500) clamp load addresses (stay in-bounds, uniform EXEC)
//     and skip only the store.
// ---------------------------------------------------------------------------
__global__ __launch_bounds__(256) void fc_gemm(const _Float16* __restrict__ hs,
                                               const _Float16* __restrict__ fcw,
                                               const float* __restrict__ fcb,
                                               float* __restrict__ out) {
  int wave = threadIdx.x >> 5, lane = threadIdx.x & 31;
  int hf = lane >> 4, l15 = lane & 15;
  int mB = blockIdx.x * 16;
  int tileBase = blockIdx.y * 32 + wave * 4;

  int nBc[4];
#pragma unroll
  for (int t = 0; t < 4; ++t) {
    int tile = tileBase + t;
    nBc[t] = (tile < 500 ? tile : 499) * 16;      // clamped load tile
  }

  v8f acc[4] = {};
  v16h aCur = load_a_frag(hs, 256, mB, 0, lane);
  v16h bCur[4];
#pragma unroll
  for (int t = 0; t < 4; ++t) bCur[t] = load_b_frag(fcw, 256, nBc[t], 0, lane);

#pragma unroll
  for (int k = 0; k < 8; ++k) {
    v16h aNxt;
    v16h bNxt[4];
    if (k < 7) {                                  // prefetch k+1 (compile-time)
      aNxt = load_a_frag(hs, 256, mB, (k + 1) * 32, lane);
#pragma unroll
      for (int t = 0; t < 4; ++t)
        bNxt[t] = load_b_frag(fcw, 256, nBc[t], (k + 1) * 32, lane);
    }
#pragma unroll
    for (int t = 0; t < 4; ++t) acc[t] = wmma_f16(aCur, bCur[t], acc[t]);
    if (k < 7) {
      aCur = aNxt;
#pragma unroll
      for (int t = 0; t < 4; ++t) bCur[t] = bNxt[t];
    }
  }

#pragma unroll
  for (int t = 0; t < 4; ++t) {
    int tile = tileBase + t;
    if (tile < 500) {                             // wave-uniform store guard
      int n = tile * 16 + l15;
      float bias = fcb[n];
#pragma unroll
      for (int v = 0; v < 8; ++v) {
        size_t m = (size_t)(mB + v + hf * 8);
        out[m * 8000 + n] = acc[t][v] + bias;
      }
    }
  }
}

// ---------------------------------------------------------------------------
// host launcher
// ---------------------------------------------------------------------------
extern "C" void kernel_launch(void* const* d_in, const int* in_sizes, int n_in,
                              void* d_out, int out_size, void* d_ws, size_t ws_size,
                              hipStream_t stream) {
  const long long* x   = (const long long*)d_in[0];   // (16,1024,2) int64
  const float* embed_x = (const float*)d_in[1];       // (8000,128)
  const float* embed_l = (const float*)d_in[2];       // (100,128)
  const float* W_ih    = (const float*)d_in[3];       // (1024,256)
  const float* W_hh    = (const float*)d_in[4];       // (1024,256)
  const float* b_ih    = (const float*)d_in[5];       // (1024)
  const float* b_hh    = (const float*)d_in[6];       // (1024)
  const float* fc_w    = (const float*)d_in[7];       // (8000,256)
  const float* fc_b    = (const float*)d_in[8];       // (8000)
  float* out = (float*)d_out;                         // (16,1024,8000)

  char* ws = (char*)d_ws;
  float*    xproj   = (float*)   (ws);                         // 64 MB
  _Float16* inp16   = (_Float16*)(ws + (size_t)67108864);      //  8 MB
  _Float16* hs16    = (_Float16*)(ws + (size_t)75497472);      //  8 MB
  _Float16* wih16   = (_Float16*)(ws + (size_t)83886080);      // .5 MB
  _Float16* whh16   = (_Float16*)(ws + (size_t)84410368);      // .5 MB
  _Float16* fcw16   = (_Float16*)(ws + (size_t)84934656);      //  4 MB

  // K1: weight converts
  cvt_f32_f16<<<512, 256, 0, stream>>>(W_ih, wih16, 1024 * 256);
  cvt_f32_f16<<<512, 256, 0, stream>>>(W_hh, whh16, 1024 * 256);
  cvt_f32_f16<<<2048, 256, 0, stream>>>(fc_w, fcw16, 8000 * 256);

  // K2: embedding gather + concat
  embed_gather<<<16384, 256, 0, stream>>>(x, embed_x, embed_l, inp16);

  // K3: input projection (WMMA, pipelined)
  in_proj<<<dim3(1024, 2), 256, 0, stream>>>(inp16, wih16, b_ih, b_hh, xproj);

  // K4: sequential LSTM scan (persistent single block, WMMA + LDS)
  lstm_scan<<<1, 256, 0, stream>>>(xproj, whh16, hs16);

  // K5: big FC GEMM (WMMA, pipelined) + bias
  fc_gemm<<<dim3(1024, 16), 256, 0, stream>>>(hs16, fcw16, fc_b, out);
}
